// MultiHeadAttention_11252814316111
// MI455X (gfx1250) — matrix operand order repaired and checked by reading, not compiled, not run on hardware
//
#include <hip/hip_runtime.h>

typedef __attribute__((ext_vector_type(16))) _Float16 v16h;
typedef __attribute__((ext_vector_type(8)))  float    v8f;

#define WMMA32F16(A_, B_, C_) \
  __builtin_amdgcn_wmma_f32_16x16x32_f16(false, (A_), false, (B_), (short)0, (C_), false, false)

// CDNA5 async global->LDS DMA (ASYNCcnt-tracked), if the toolchain exposes it.
#if defined(__has_builtin)
#if __has_builtin(__builtin_amdgcn_global_load_async_to_lds_b128)
#define HAS_ASYNC_LDS 1
#endif
#endif
#ifndef HAS_ASYNC_LDS
#define HAS_ASYNC_LDS 0
#endif

// Types matching the builtin's parameters exactly:
//   param0: __attribute__((vector_size(16))) int  addrspace(1)*   (global src)
//   param1: __attribute__((vector_size(16))) int  addrspace(3)*   (LDS dst)
typedef int gv4i __attribute__((vector_size(16)));
typedef __attribute__((address_space(1))) gv4i as1_v4i;
typedef __attribute__((address_space(3))) gv4i as3_v4i;

constexpr int kB   = 4;
constexpr int kS   = 2048;
constexpr int kHID = 1024;
constexpr int kH   = 16;
constexpr int kD   = 64;

// ---------------------------------------------------------------------------
// WMMA fragment loaders, wave32; contiguous k order (k = 16*(lane/16) + e) in A and B
// ---------------------------------------------------------------------------

// A-matrix 16x32 (MxK) f16 from row-major LDS tile.
__device__ __forceinline__ v16h load_a_frag(const _Float16* base, int ld) {
  const int lane = threadIdx.x & 31;
  const int row  = lane & 15;
  const int koff = (lane & 16) ? 16 : 0;
  v16h a;
#pragma unroll
  for (int j = 0; j < 8; ++j) {
    const int k = koff + 2 * j;
    a[2 * j]     = base[row * ld + k];
    a[2 * j + 1] = base[row * ld + k + 1];
  }
  return a;
}

// B-matrix 32x16 (KxN): element (k,n) = src[n*ld + k]   ("B = src^T")
__device__ __forceinline__ v16h load_b_frag_nt(const _Float16* src, int ld) {
  const int lane = threadIdx.x & 31;
  const int n    = lane & 15;
  const int koff = (lane & 16) ? 16 : 0;
  v16h b;
#pragma unroll
  for (int e = 0; e < 16; ++e) b[e] = src[n * ld + koff + e];
  return b;
}

// B-matrix 32x16 (KxN): element (k,n) = src[k*ld + n]
__device__ __forceinline__ v16h load_b_frag_nn(const _Float16* src, int ld) {
  const int lane = threadIdx.x & 31;
  const int n    = lane & 15;
  const int koff = (lane & 16) ? 16 : 0;
  v16h b;
#pragma unroll
  for (int e = 0; e < 16; ++e) b[e] = src[(koff + e) * ld + n];
  return b;
}

// row-wise reductions of C-layout tiles (16-lane half-wave)
__device__ __forceinline__ float redmax16(float v) {
  v = fmaxf(v, __shfl_xor(v, 8, 32));
  v = fmaxf(v, __shfl_xor(v, 4, 32));
  v = fmaxf(v, __shfl_xor(v, 2, 32));
  v = fmaxf(v, __shfl_xor(v, 1, 32));
  return v;
}
__device__ __forceinline__ float redsum16(float v) {
  v += __shfl_xor(v, 8, 32);
  v += __shfl_xor(v, 4, 32);
  v += __shfl_xor(v, 2, 32);
  v += __shfl_xor(v, 1, 32);
  return v;
}

// ---------------------------------------------------------------------------
// GEMM: out = A @ W^T + bias, double-buffered LDS, 1 barrier / K-step.
// MODE 0: A = fp32 activations, out = f16 in [B,H,S,D] split-head layout
// MODE 1: A = f16 ctx,          out = fp32 plain [M,N]
// Block 256 thr (8 waves), tile 128x128x32; wave = 64x32 patch (4x2 WMMA tiles)
// ---------------------------------------------------------------------------
template <int MODE>
__global__ __launch_bounds__(256) void gemm_wmma(const void* __restrict__ Ap,
                                                 const float* __restrict__ W,
                                                 const float* __restrict__ bias,
                                                 void* __restrict__ outp) {
  __shared__ alignas(16) _Float16 sA[2][128 * 32];
  __shared__ alignas(16) _Float16 sW[2][128 * 32];
  const int tid  = threadIdx.x;
  const int lane = tid & 31;
  const int wid  = tid >> 5;
  const int wm   = wid >> 2;  // 0..1
  const int wn   = wid & 3;   // 0..3
  const int m0   = blockIdx.y * 128;
  const int n0   = blockIdx.x * 128;

  v8f acc[4][2];
#pragma unroll
  for (int i = 0; i < 4; ++i)
#pragma unroll
    for (int j = 0; j < 2; ++j)
      acc[i][j] = (v8f){0.f, 0.f, 0.f, 0.f, 0.f, 0.f, 0.f, 0.f};

  float4 rA[4];  // MODE 0 staging
  uint4  rAh[2]; // MODE 1 staging
  float4 rW[4];

  auto gload = [&](int k0) {
    if constexpr (MODE == 0) {
      const float* A = (const float*)Ap;
#pragma unroll
      for (int i = 0; i < 4; ++i) {
        const int idx = (tid + i * 256) * 4, row = idx >> 5, col = idx & 31;
        rA[i] = *(const float4*)(A + (size_t)(m0 + row) * kHID + k0 + col);
      }
    } else {
      const _Float16* A = (const _Float16*)Ap;
#pragma unroll
      for (int i = 0; i < 2; ++i) {
        const int idx = (tid + i * 256) * 8, row = idx >> 5, col = idx & 31;
        rAh[i] = *(const uint4*)(A + (size_t)(m0 + row) * kHID + k0 + col);
      }
    }
#pragma unroll
    for (int i = 0; i < 4; ++i) {
      const int idx = (tid + i * 256) * 4, row = idx >> 5, col = idx & 31;
      rW[i] = *(const float4*)(W + (size_t)(n0 + row) * kHID + k0 + col);
    }
  };
  auto sstore = [&](int buf) {
    if constexpr (MODE == 0) {
#pragma unroll
      for (int i = 0; i < 4; ++i) {
        const int idx = (tid + i * 256) * 4, row = idx >> 5, col = idx & 31;
        _Float16* p = &sA[buf][row * 32 + col];
        p[0] = (_Float16)rA[i].x; p[1] = (_Float16)rA[i].y;
        p[2] = (_Float16)rA[i].z; p[3] = (_Float16)rA[i].w;
      }
    } else {
#pragma unroll
      for (int i = 0; i < 2; ++i) {
        const int idx = (tid + i * 256) * 8, row = idx >> 5, col = idx & 31;
        *(uint4*)&sA[buf][row * 32 + col] = rAh[i];
      }
    }
#pragma unroll
    for (int i = 0; i < 4; ++i) {
      const int idx = (tid + i * 256) * 4, row = idx >> 5, col = idx & 31;
      _Float16* p = &sW[buf][row * 32 + col];
      p[0] = (_Float16)rW[i].x; p[1] = (_Float16)rW[i].y;
      p[2] = (_Float16)rW[i].z; p[3] = (_Float16)rW[i].w;
    }
  };

  constexpr int NK = kHID / 32;  // 32 K-steps
  gload(0);
  sstore(0);
  __syncthreads();

  for (int kk = 0; kk < NK; ++kk) {
    const int cur = kk & 1;
    if (kk + 1 < NK) gload((kk + 1) * 32);  // overlap next tile with compute

    v16h af[4], bf[2];
#pragma unroll
    for (int mi = 0; mi < 4; ++mi)
      af[mi] = load_a_frag(&sA[cur][(wm * 64 + mi * 16) * 32], 32);
#pragma unroll
    for (int ni = 0; ni < 2; ++ni)
      bf[ni] = load_b_frag_nt(&sW[cur][(wn * 32 + ni * 16) * 32], 32);
#pragma unroll
    for (int mi = 0; mi < 4; ++mi)
#pragma unroll
      for (int ni = 0; ni < 2; ++ni)
        acc[mi][ni] = WMMA32F16(af[mi], bf[ni], acc[mi][ni]);

    if (kk + 1 < NK) sstore(cur ^ 1);
    __syncthreads();
  }

  // epilogue: C layout = VGPR r -> row (r | r+8), col = lane&15
  const int mh = (lane & 16) ? 8 : 0;
  const int nl = lane & 15;
#pragma unroll
  for (int mi = 0; mi < 4; ++mi)
#pragma unroll
    for (int ni = 0; ni < 2; ++ni)
#pragma unroll
      for (int r = 0; r < 8; ++r) {
        const int M = m0 + wm * 64 + mi * 16 + mh + r;
        const int N = n0 + wn * 32 + ni * 16 + nl;
        const float v = acc[mi][ni][r] + bias[N];
        if constexpr (MODE == 0) {
          _Float16* out = (_Float16*)outp;
          const int b = M >> 11, s = M & (kS - 1);
          const int h = N >> 6, d = N & (kD - 1);
          out[(((size_t)(b * kH + h)) * kS + s) * kD + d] = (_Float16)v;
        } else {
          float* out = (float*)outp;
          out[(size_t)M * kHID + N] = v;
        }
      }
}

// ---------------------------------------------------------------------------
// Flash attention, causal. Grid (S/128, H, B), block 256 (8 waves).
// Each wave owns 16 query rows; K/V streamed in 32-key chunks, double-buffered
// (async global->LDS DMA when available).
// ---------------------------------------------------------------------------
__global__ __launch_bounds__(256) void flash_attn(const _Float16* __restrict__ qh,
                                                  const _Float16* __restrict__ kh,
                                                  const _Float16* __restrict__ vh,
                                                  _Float16* __restrict__ ctx) {
  __shared__ alignas(16) _Float16 sQ[128 * 64];
  __shared__ alignas(16) _Float16 sK[2][32 * 64];
  __shared__ alignas(16) _Float16 sV[2][32 * 64];
  __shared__ alignas(16) _Float16 sP[8][16 * 32];

  const int tid  = threadIdx.x;
  const int lane = tid & 31;
  const int wid  = tid >> 5;
  const int q0   = blockIdx.x * 128;
  const int h    = blockIdx.y;
  const int bidx = blockIdx.z;

  const size_t bh = (size_t)(bidx * kH + h) * kS * kD;
  const _Float16* Q = qh + bh;
  const _Float16* K = kh + bh;
  const _Float16* V = vh + bh;

  // stage Q tile (128 x 64 f16) into LDS
#pragma unroll
  for (int i = 0; i < 4; ++i) {
    const int idx = (tid + i * 256) * 8;
    const int row = idx >> 6, col = idx & 63;
    *(uint4*)&sQ[idx] = *(const uint4*)(Q + (size_t)(q0 + row) * kD + col);
  }
  __syncthreads();

  const int qw = q0 + wid * 16;
  const v16h aq0 = load_a_frag(&sQ[(wid * 16) * kD], kD);       // dims 0..31
  const v16h aq1 = load_a_frag(&sQ[(wid * 16) * kD + 32], kD);  // dims 32..63

  const int mh = (lane & 16) ? 8 : 0;
  const int nl = lane & 15;

  float mrow[8], lrow[8];
  v8f accO[4];
#pragma unroll
  for (int r = 0; r < 8; ++r) { mrow[r] = -1e30f; lrow[r] = 0.f; }
#pragma unroll
  for (int j = 0; j < 4; ++j)
    accO[j] = (v8f){0.f, 0.f, 0.f, 0.f, 0.f, 0.f, 0.f, 0.f};

  const int nchunks = (q0 >> 5) + 4;
  const float scale = 0.125f;  // 1/sqrt(64)

  // per-thread chunk addressing: one 16-byte slice of K and V each
  const int cidx = tid * 8;
  const int crow = cidx >> 6, ccol = cidx & 63;

#if HAS_ASYNC_LDS
  auto issue_chunk = [&](int c, int buf) {
    const size_t g = ((size_t)(c * 32 + crow)) * kD + ccol;
    __builtin_amdgcn_global_load_async_to_lds_b128((as1_v4i*)(K + g),
                                                   (as3_v4i*)&sK[buf][cidx], 0, 0);
    __builtin_amdgcn_global_load_async_to_lds_b128((as1_v4i*)(V + g),
                                                   (as3_v4i*)&sV[buf][cidx], 0, 0);
  };
  issue_chunk(0, 0);  // ASYNCcnt = 2 per wave
#else
  uint4 rk, rv;
  auto load_chunk = [&](int c) {
    const size_t g = ((size_t)(c * 32 + crow)) * kD + ccol;
    rk = *(const uint4*)(K + g);
    rv = *(const uint4*)(V + g);
  };
  auto store_chunk = [&](int buf) {
    *(uint4*)&sK[buf][cidx] = rk;
    *(uint4*)&sV[buf][cidx] = rv;
  };
  load_chunk(0);
  store_chunk(0);
  __syncthreads();
#endif

  for (int c = 0; c < nchunks; ++c) {
    const int cur = c & 1;
#if HAS_ASYNC_LDS
    __syncthreads();  // all waves done reading buf cur^1 (iter c-1)
    if (c + 1 < nchunks) {
      issue_chunk(c + 1, cur ^ 1);                       // DMA next chunk
      asm volatile("s_wait_asynccnt 2" ::: "memory");    // chunk c landed
    } else {
      asm volatile("s_wait_asynccnt 0" ::: "memory");
    }
    __syncthreads();  // chunk c visible to every wave
#else
    if (c + 1 < nchunks) {
      load_chunk(c + 1);  // overlap global load with compute
      const size_t nidx = ((size_t)((c + 1) * 32) + crow) * kD + ccol;
      __builtin_prefetch(K + nidx, 0, 3);
      __builtin_prefetch(V + nidx, 0, 3);
    }
#endif

    if (c * 32 <= qw + 15) {  // wave-uniform causal skip
      // S = Q K^T : two 16x16 key tiles
      v8f s0 = (v8f){0.f, 0.f, 0.f, 0.f, 0.f, 0.f, 0.f, 0.f};
      v8f s1 = s0;
      s0 = WMMA32F16(aq0, load_b_frag_nt(&sK[cur][0], kD), s0);
      s0 = WMMA32F16(aq1, load_b_frag_nt(&sK[cur][32], kD), s0);
      s1 = WMMA32F16(aq0, load_b_frag_nt(&sK[cur][16 * kD], kD), s1);
      s1 = WMMA32F16(aq1, load_b_frag_nt(&sK[cur][16 * kD + 32], kD), s1);

      // online softmax on C-layout registers
#pragma unroll
      for (int r = 0; r < 8; ++r) {
        const int qi = qw + mh + r;
        float x0 = s0[r] * scale;
        float x1 = s1[r] * scale;
        if (c * 32 + nl > qi)      x0 = -1e30f;
        if (c * 32 + 16 + nl > qi) x1 = -1e30f;
        const float t  = redmax16(fmaxf(x0, x1));
        const float mn = fmaxf(mrow[r], t);
        const float alpha = __expf(mrow[r] - mn);
        mrow[r] = mn;
        const float p0 = __expf(x0 - mn);
        const float p1 = __expf(x1 - mn);
        lrow[r] = lrow[r] * alpha + redsum16(p0 + p1);
#pragma unroll
        for (int j = 0; j < 4; ++j) accO[j][r] *= alpha;
        const int M = mh + r;
        sP[wid][M * 32 + nl]      = (_Float16)p0;
        sP[wid][M * 32 + 16 + nl] = (_Float16)p1;
      }
      __builtin_amdgcn_wave_barrier();  // wave-private LDS round trip; DS in-order
      asm volatile("" ::: "memory");

      const v16h ap = load_a_frag(sP[wid], 32);  // P as 16x32 A-matrix
#pragma unroll
      for (int dt = 0; dt < 4; ++dt)
        accO[dt] = WMMA32F16(ap, load_b_frag_nn(&sV[cur][dt * 16], kD), accO[dt]);
    }

#if !HAS_ASYNC_LDS
    if (c + 1 < nchunks) store_chunk(cur ^ 1);
    __syncthreads();
#endif
  }

  // normalize; stage wave's 16x64 output tile in (now dead) sQ, then b128 stores
  _Float16* sO = &sQ[wid * 16 * kD];
#pragma unroll
  for (int dt = 0; dt < 4; ++dt)
#pragma unroll
    for (int r = 0; r < 8; ++r) {
      const float o = accO[dt][r] / lrow[r];
      sO[(mh + r) * kD + dt * 16 + nl] = (_Float16)o;
    }
  __builtin_amdgcn_wave_barrier();
  asm volatile("" ::: "memory");
#pragma unroll
  for (int j = 0; j < 4; ++j) {
    const int t   = lane + j * 32;
    const int row = t >> 3;
    const int col = (t & 7) * 8;
    *(uint4*)(ctx + ((size_t)bidx * kS + qw + row) * kHID + h * kD + col) =
        *(const uint4*)&sO[row * kD + col];
  }
}

// ---------------------------------------------------------------------------
extern "C" void kernel_launch(void* const* d_in, const int* in_sizes, int n_in,
                              void* d_out, int out_size, void* d_ws, size_t ws_size,
                              hipStream_t stream) {
  (void)in_sizes; (void)n_in; (void)out_size; (void)ws_size;

  const float* q  = (const float*)d_in[0];
  const float* k  = (const float*)d_in[1];
  const float* v  = (const float*)d_in[2];
  // d_in[3] = attn_mask (causal) — computed analytically in-kernel
  const float* Wq = (const float*)d_in[4];
  const float* bq = (const float*)d_in[5];
  const float* Wk = (const float*)d_in[6];
  const float* bk = (const float*)d_in[7];
  const float* Wv = (const float*)d_in[8];
  const float* bv = (const float*)d_in[9];
  const float* Wo = (const float*)d_in[10];
  const float* bo = (const float*)d_in[11];

  const size_t nElem = (size_t)kB * kS * kHID;  // 8,388,608 (16 MB f16 each)
  _Float16* qh  = (_Float16*)d_ws;
  _Float16* kh  = qh + nElem;
  _Float16* vh  = kh + nElem;
  _Float16* ctx = vh + nElem;

  const dim3 gemm_grid(kHID / 128, (kB * kS) / 128);  // (8, 64)
  gemm_wmma<0><<<gemm_grid, 256, 0, stream>>>(q, Wq, bq, qh);
  gemm_wmma<0><<<gemm_grid, 256, 0, stream>>>(k, Wk, bk, kh);
  gemm_wmma<0><<<gemm_grid, 256, 0, stream>>>(v, Wv, bv, vh);

  flash_attn<<<dim3(kS / 128, kH, kB), 256, 0, stream>>>(qh, kh, vh, ctx);

  gemm_wmma<1><<<gemm_grid, 256, 0, stream>>>(ctx, Wo, bo, (float*)d_out);
}
